// LayerStacks_5849745457168
// MI455X (gfx1250) — matrix operand order repaired and checked by reading, not compiled, not run on hardware
//
#include <hip/hip_runtime.h>
#include <hip/hip_bf16.h>

typedef __attribute__((ext_vector_type(4)))  _Float16 v4h;
typedef __attribute__((ext_vector_type(8)))  _Float16 v8h;
typedef __attribute__((ext_vector_type(16))) _Float16 v16h;
typedef __attribute__((ext_vector_type(8)))  float    v8f;

#define BATCH   65536
#define XS      520   // padded halfs per sample row in LDS (1040B, conflict-free)

__device__ __forceinline__ v16h mk16(v8h lo, v8h hi) {
    return __builtin_shufflevector(lo, hi, 0,1,2,3,4,5,6,7,8,9,10,11,12,13,14,15);
}

// ---------------- prep: fold factorized weights, convert to f16 ----------------
__global__ void ls_prep(const float* __restrict__ l1w, const float* __restrict__ l1f,
                        const float* __restrict__ l2w, const float* __restrict__ l2f,
                        const float* __restrict__ ow,  const float* __restrict__ ofw,
                        _Float16* __restrict__ W1h, _Float16* __restrict__ W2h,
                        float* __restrict__ W3) {
    int i = blockIdx.x * blockDim.x + threadIdx.x;
    if (i < 131072) {                       // [256 rows][512]  row = e*32+r
        int row = i >> 9, c = i & 511;
        W1h[i] = (_Float16)(l1w[i] + l1f[(row & 31) * 512 + c]);
    }
    int j = i - 131072;
    if (j >= 0 && j < 8192) {               // [256 rows][32]
        int row = j >> 5, c = j & 31;
        W2h[j] = (_Float16)(l2w[j] + l2f[(row & 31) * 32 + c]);
    }
    int m = i - (131072 + 8192);
    if (m >= 0 && m < 256) {                // [8][32]
        W3[m] = ow[m] + ofw[m & 31];
    }
}

// ---------------- binning ----------------
__global__ void ls_zero(int* __restrict__ cnt) {
    if (threadIdx.x < 8) cnt[threadIdx.x] = 0;
}

__global__ void ls_bin(const int* __restrict__ lsidx, int* __restrict__ cnt,
                       int* __restrict__ perm) {
    int i = blockIdx.x * blockDim.x + threadIdx.x;
    if (i < BATCH) {
        int e = lsidx[i] & 7;
        int pos = atomicAdd(&cnt[e], 1);
        perm[e * BATCH + pos] = i;
    }
}

// ---------------- main: 16 same-stack samples per block, WMMA f16->f32 ----------------
__global__ __launch_bounds__(64) void ls_main(
    const float*    __restrict__ x,
    const _Float16* __restrict__ W1h,   // [8][32][512]
    const _Float16* __restrict__ W2h,   // [8][32][32]
    const float*    __restrict__ W3,    // [8][32]
    const float*    __restrict__ b1,    // [256]
    const float*    __restrict__ b2,    // [256]
    const float*    __restrict__ b3,    // [8]
    const int*      __restrict__ cnt,
    const int*      __restrict__ perm,
    float*          __restrict__ out) {

    const int e    = blockIdx.y;
    const int tile = blockIdx.x;
    const int n    = cnt[e];
    if (tile * 16 >= n) return;                      // uniform early-exit
    const int tbase = tile * 16;
    int rem = n - tbase; if (rem > 16) rem = 16;

    __shared__ _Float16 xh[16 * XS];                 // x tile, f16, [sample][k]
    __shared__ _Float16 y1t[16 * 40];                // stage1 act, [sample][row]
    __shared__ float    y2l[16 * 33];                // stage2 act, [sample][row]
    __shared__ int      smp[16];

    const int tid  = threadIdx.x;
    const int wave = tid >> 5;
    const int lane = tid & 31;

    if (tid < 16) {
        int j = (tid < rem) ? tid : 0;               // pad partial tile w/ sample 0
        smp[tid] = perm[e * BATCH + tbase + j];
    }
    __syncthreads();

    // ---- stream x tile (fp32, coalesced b128) -> convert -> LDS f16 ----
    for (int t = tid; t < 16 * 128; t += 64) {
        int j = t >> 7, c4 = t & 127;
        float4 f = ((const float4*)x)[smp[j] * 128 + c4];
        *(v4h*)(&xh[j * XS + c4 * 4]) =
            (v4h){(_Float16)f.x, (_Float16)f.y, (_Float16)f.z, (_Float16)f.w};
    }
    __syncthreads();

    // ---- stage 1: D[16 rows x 16 samples] += W1c(16x32) x X(32x16), K=512 ----
    // A as loaded here (contiguous k, matches B): lanes 0-15 row M, K {k..k+15}; lanes 16-31 K {k+16..k+31}
    // B as loaded here (contiguous k): lanes 0-15 col N, K {k..k+15}; lanes 16-31 K {k+16..k+31}
    const _Float16* wrow = W1h + (size_t)(e * 32 + wave * 16 + (lane & 15)) * 512;
    const int khalf = (lane >> 4) << 4;                          // 0 or 16
    const _Float16* xrow = xh + (lane & 15) * XS + ((lane >> 4) << 4);  // +16 halfs

    v8f acc = {};
#pragma unroll
    for (int k = 0; k < 512; k += 32) {
        v16h a = mk16(*(const v8h*)(wrow + k + khalf), *(const v8h*)(wrow + k + khalf + 8));
        v16h b = mk16(*(const v8h*)(xrow + k),          *(const v8h*)(xrow + k + 8));
        acc = __builtin_amdgcn_wmma_f32_16x16x32_f16(false, a, false, b,
                                                     (short)0, acc, false, false);
    }

    // bias + clamp, hand off transposed as f16 (rows = K of stage 2)
    {
        int rl  = (lane >> 4) << 3;
        int col = lane & 15;
#pragma unroll
        for (int vr = 0; vr < 8; ++vr) {
            int row = wave * 16 + rl + vr;
            float v = fminf(fmaxf(acc[vr] + b1[e * 32 + row], 0.f), 1.f);
            y1t[col * 40 + row] = (_Float16)v;
        }
    }
    __syncthreads();

    // ---- stage 2: D[16x16] = W2c(16x32) x Y1(32x16), single K=32 step ----
    const _Float16* w2row = W2h + (size_t)(e * 32 + wave * 16 + (lane & 15)) * 32;
    const _Float16* yrow  = y1t + (lane & 15) * 40 + ((lane >> 4) << 4);
    v8f acc2 = {};
    {
        v16h a = mk16(*(const v8h*)(w2row + khalf), *(const v8h*)(w2row + khalf + 8));
        v16h b = mk16(*(const v8h*)(yrow),          *(const v8h*)(yrow + 8));
        acc2 = __builtin_amdgcn_wmma_f32_16x16x32_f16(false, a, false, b,
                                                      (short)0, acc2, false, false);
    }
    {
        int rl  = (lane >> 4) << 3;
        int col = lane & 15;
#pragma unroll
        for (int vr = 0; vr < 8; ++vr) {
            int row = wave * 16 + rl + vr;
            float v = fminf(fmaxf(acc2[vr] + b2[e * 32 + row], 0.f), 1.f);
            y2l[col * 33 + row] = v;
        }
    }
    __syncthreads();

    // ---- output stage: scalar per sample ----
    if (tid < rem) {
        float a = b3[e];
        const float* w3 = W3 + e * 32;
        const float* y  = y2l + tid * 33;
#pragma unroll
        for (int r = 0; r < 32; ++r) a += w3[r] * y[r];
        out[smp[tid]] = a;
    }
}

extern "C" void kernel_launch(void* const* d_in, const int* in_sizes, int n_in,
                              void* d_out, int out_size, void* d_ws, size_t ws_size,
                              hipStream_t stream) {
    (void)in_sizes; (void)n_in; (void)out_size; (void)ws_size;
    const float* x    = (const float*)d_in[0];
    const int*   lsi  = (const int*)  d_in[1];
    const float* l1w  = (const float*)d_in[2];
    const float* l1b  = (const float*)d_in[3];
    const float* l1fw = (const float*)d_in[4];
    const float* l2w  = (const float*)d_in[5];
    const float* l2b  = (const float*)d_in[6];
    const float* l2fw = (const float*)d_in[7];
    const float* ow   = (const float*)d_in[8];
    const float* ob   = (const float*)d_in[9];
    const float* ofw  = (const float*)d_in[10];
    float* out = (float*)d_out;

    char* ws = (char*)d_ws;
    _Float16* W1h  = (_Float16*)(ws + 0);        // 262144 B
    _Float16* W2h  = (_Float16*)(ws + 262144);   //  16384 B
    float*    W3   = (float*)   (ws + 278528);   //   1024 B
    int*      cntv = (int*)     (ws + 279552);   //     32 B (+pad)
    int*      perm = (int*)     (ws + 279680);   // 2 MiB

    ls_prep<<<(131072 + 8192 + 256 + 255) / 256, 256, 0, stream>>>(
        l1w, l1fw, l2w, l2fw, ow, ofw, W1h, W2h, W3);
    ls_zero<<<1, 32, 0, stream>>>(cntv);
    ls_bin<<<BATCH / 256, 256, 0, stream>>>(lsi, cntv, perm);
    ls_main<<<dim3(BATCH / 16, 8), 64, 0, stream>>>(
        x, W1h, W2h, W3, l1b, l2b, ob, cntv, perm, out);
}
